// DenseFeatureExtractionModule_AP_OS2_82815559401759
// MI455X (gfx1250) — matrix operand order repaired and checked by reading, not compiled, not run on hardware
//
#include <hip/hip_runtime.h>

typedef unsigned short u16;
typedef __attribute__((ext_vector_type(16))) __bf16 v16bf;
typedef __attribute__((ext_vector_type(8)))  float  v8f;
typedef __attribute__((ext_vector_type(4)))  int    v4i;

union FragU { v16bf v; uint4 q[2]; };

__device__ __forceinline__ u16 f2bf(float f) {
  union { float f; unsigned u; } c; c.f = f;
  unsigned u = c.u + 0x7FFFu + ((c.u >> 16) & 1u);   // RNE truncate to bf16
  return (u16)(u >> 16);
}
__device__ __forceinline__ float bf2f(u16 h) {
  union { unsigned u; float f; } c; c.u = ((unsigned)h) << 16;
  return c.f;
}

// ---------------------------------------------------------------- fill zero
__global__ void fill_zero(uint4* __restrict__ p, size_t n16) {
  size_t i = (size_t)blockIdx.x * blockDim.x + threadIdx.x;
  size_t stride = (size_t)gridDim.x * blockDim.x;
  uint4 z; z.x = z.y = z.z = z.w = 0u;
  for (; i < n16; i += stride) p[i] = z;
}

// ---------------------------------------------------------------- mask expand
// pooling_mask [B,1,48,48] int -> mask8 [B,192,192] u8 (repeat 4x4)
__global__ void expand_mask(const int* __restrict__ pm, unsigned char* __restrict__ m8) {
  int idx = blockIdx.x * blockDim.x + threadIdx.x;
  const int total = 2 * 192 * 192;
  if (idx >= total) return;
  int x = idx % 192, y = (idx / 192) % 192, b = idx / (192 * 192);
  m8[idx] = (unsigned char)(pm[((size_t)b * 48 + (y >> 2)) * 48 + (x >> 2)] != 0);
}

// ---------------------------------------------------------------- weight pack
// OIHW f32 -> fragment-major bf16: frag f = (tap*KC+kc)*NT+nt, 32 lanes x 16 vals.
// B-matrix 32x16 bf16 layout: lane L holds column N=L%16, K = (L/16)*8 + s (s<8), 16 + (L/16)*8 + (s-8) (s>=8).
__global__ void pack_w(const float* __restrict__ w, u16* __restrict__ wp,
                       int Cin, int Cout) {
  const int KC = Cin >> 5, NT = Cout >> 4;
  size_t nel = (size_t)9 * Cin * Cout;
  size_t idx = (size_t)blockIdx.x * blockDim.x + threadIdx.x;
  if (idx >= nel) return;
  int s    = (int)(idx & 15);
  int lane = (int)((idx >> 4) & 31);
  size_t f = idx >> 9;
  int nt  = (int)(f % NT); f /= NT;
  int kc  = (int)(f % KC);
  int tap = (int)(f / KC);
  int k_local = (s < 8) ? (lane >> 4) * 8 + s : 16 + (lane >> 4) * 8 + (s - 8);
  int cin  = kc * 32 + k_local;
  int cout = nt * 16 + (lane & 15);
  wp[idx] = f2bf(w[((size_t)cout * Cin + cin) * 9 + tap]);
}

// ---------------------------------------------------------------- conv1 direct
// x [2,3,384,384] f32 NCHW -> out NHWC bf16 padded P=1 [2,386,386,64], ReLU
__global__ void conv1_direct(const float* __restrict__ x, const float* __restrict__ w,
                             const float* __restrict__ bias, u16* __restrict__ out) {
  const int H = 384, W = 384, Hp = 386, Wp = 386;
  size_t idx = (size_t)blockIdx.x * blockDim.x + threadIdx.x;
  size_t total = (size_t)2 * H * W * 64;
  if (idx >= total) return;
  int co = (int)(idx & 63);
  size_t t = idx >> 6;
  int px = (int)(t % W); t /= W;
  int py = (int)(t % H); int b = (int)(t / H);
  float sum = bias[co];
  for (int ci = 0; ci < 3; ++ci)
    for (int ky = 0; ky < 3; ++ky) {
      int iy = py + ky - 1;
      if (iy < 0 || iy >= H) continue;
      for (int kx = 0; kx < 3; ++kx) {
        int ix = px + kx - 1;
        if (ix < 0 || ix >= W) continue;
        sum += x[(((size_t)b * 3 + ci) * H + iy) * W + ix] *
               w[((co * 3 + ci) * 3 + ky) * 3 + kx];
      }
    }
  sum = sum > 0.f ? sum : 0.f;
  out[(((size_t)b * Hp + py + 1) * Wp + px + 1) * 64 + co] = f2bf(sum);
}

// ---------------------------------------------------------------- maxpool 2x2
// in [2,386,386,64] (P=1, logical 384) -> out [2,200,200,64] (P=4, logical 192)
__global__ void maxpool2(const u16* __restrict__ in, u16* __restrict__ out) {
  size_t idx = (size_t)blockIdx.x * blockDim.x + threadIdx.x;
  size_t total = (size_t)2 * 192 * 192 * 64;
  if (idx >= total) return;
  int c = (int)(idx & 63);
  size_t t = idx >> 6;
  int x = (int)(t % 192); t /= 192;
  int y = (int)(t % 192); int b = (int)(t / 192);
  float v = -3.4e38f;
  for (int dy = 0; dy < 2; ++dy)
    for (int dx = 0; dx < 2; ++dx) {
      float u = bf2f(in[(((size_t)b * 386 + 2 * y + dy + 1) * 386 + 2 * x + dx + 1) * 64 + c]);
      v = fmaxf(v, u);
    }
  out[(((size_t)b * 200 + y + 4) * 200 + x + 4) * 64 + c] = f2bf(v);
}

// ---------------------------------------------------------------- irregular pool
// block-max over s x s replicated, only where mask==1 (buffers P=4, logical 192)
__global__ void irr_pool(const u16* __restrict__ in, u16* __restrict__ out,
                         const unsigned char* __restrict__ m8, int C, int s) {
  size_t idx = (size_t)blockIdx.x * blockDim.x + threadIdx.x;
  size_t total = (size_t)2 * 192 * 192 * C;
  if (idx >= total) return;
  int c = (int)(idx % C);
  size_t t = idx / C;
  int x = (int)(t % 192); t /= 192;
  int y = (int)(t % 192); int b = (int)(t / 192);
  float v;
  if (m8[((size_t)b * 192 + y) * 192 + x]) {
    int yb = (y / s) * s, xb = (x / s) * s;
    v = -3.4e38f;
    for (int dy = 0; dy < s; ++dy)
      for (int dx = 0; dx < s; ++dx)
        v = fmaxf(v, bf2f(in[(((size_t)b * 200 + yb + dy + 4) * 200 + xb + dx + 4) * C + c]));
  } else {
    v = bf2f(in[(((size_t)b * 200 + y + 4) * 200 + x + 4) * C + c]);
  }
  out[(((size_t)b * 200 + y + 4) * 200 + x + 4) * C + c] = f2bf(v);
}

// ---------------------------------------------------------------- async->LDS
// Stage one 16-byte chunk: global (per-lane address) -> LDS (per-lane offset).
// Builtin signature (from compiler diagnostic): param0 = v4i addrspace(1)*,
// so pass int4-vector pointers with explicit address spaces.
__device__ __forceinline__ void async_g2l_b128(const uint4* gsrc, uint4* ldst,
                                               unsigned lds_byte_off) {
#if __has_builtin(__builtin_amdgcn_global_load_async_to_lds_b128)
  typedef __attribute__((address_space(1))) v4i gas_v4i;
  typedef __attribute__((address_space(3))) v4i las_v4i;
  __builtin_amdgcn_global_load_async_to_lds_b128((gas_v4i*)gsrc, (las_v4i*)ldst, 0, 0);
#else
  asm volatile("global_load_async_to_lds_b128 %0, %1, off"
               :: "v"(lds_byte_off), "v"(gsrc)
               : "memory");
#endif
}
__device__ __forceinline__ void wait_async_lds() {
#if __has_builtin(__builtin_amdgcn_s_wait_asynccnt)
  __builtin_amdgcn_s_wait_asynccnt(0);
#else
  asm volatile("s_wait_asynccnt 0x0" ::: "memory");
#endif
}

// ---------------------------------------------------------------- WMMA conv
// Implicit-GEMM 3x3 conv, NHWC bf16 padded in, fragment-packed bf16 weights.
// Block = 8 waves covering 8 consecutive y rows of one (x-tile, nt) pair, so
// all waves share one output-channel tile; B fragments for this nt are staged
// once into LDS via GLOBAL_LOAD_ASYNC_TO_LDS_B128 (ASYNCcnt) and read back
// with ds_load_b128 pairs (8x reuse).
// One wave -> 16 pixels (row y, x0..x0+15) x 16 output channels, K unrolled.
// GRAPH: second accumulator for the dilated conv, per-pixel select by mask.
// A 16x32 bf16 layout: lane L = pixel M=L%16; lanes<16 need ch {0..7,16..23},
// lanes>=16 need ch {8..15,24..31}  -> two contiguous 16B loads per fragment.
template <bool GRAPH, bool OUT_NCHW_F32, int CIN, int COUT>
__global__ __launch_bounds__(256) void conv_wmma(
    const u16* __restrict__ in, const u16* __restrict__ wp,
    const float* __restrict__ bias, const unsigned char* __restrict__ m8,
    u16* __restrict__ out_bf, float* __restrict__ out_f32,
    int H, int W, int P, int d) {
  constexpr int KC = CIN >> 5;     // K chunks of 32 channels
  constexpr int NT = COUT >> 4;    // output-channel tiles
  constexpr int NFRAG = 9 * KC;    // B fragments staged per block
  const int Wp = W + 2 * P, Hp = H + 2 * P;
  const int WT = W >> 4;

  extern __shared__ u16 lds[];     // NFRAG * 512 u16 (LDS offset 0)

  int blk = blockIdx.x;
  int nt = blk % NT; blk /= NT;
  int xt = blk % WT; blk /= WT;
  int yg = blk % (H >> 3);
  int b  = blk / (H >> 3);
  int y  = yg * 8 + (threadIdx.x >> 5);
  int lane = threadIdx.x & 31;

  // ---- async-stage B fragments for this nt into LDS (no VGPR round trip)
  {
    const uint4* src = (const uint4*)wp;
    uint4* dst = (uint4*)lds;
    for (int i = threadIdx.x; i < NFRAG * 64; i += 256) {
      int f = i >> 6, e8 = i & 63;
      size_t s = (size_t)(f * NT + nt) * 64 + e8;
      async_g2l_b128(src + s, dst + i, (unsigned)i * 16u);
    }
    wait_async_lds();               // s_wait_asynccnt 0
  }
  __syncthreads();

  int x0 = xt << 4;
  int m = lane & 15, half = lane >> 4;
  const int cA0 = half ? 8 : 0;    // + kc*32; second 16B load at +16

  v8f acc1 = {};
  v8f acc2 = {};

  #pragma unroll
  for (int ky = 0; ky < 3; ++ky) {
    #pragma unroll
    for (int kx = 0; kx < 3; ++kx) {
      const int tap = ky * 3 + kx;
      const u16* pa1 =
          in + ((size_t)(b * Hp + (y + ky - 1 + P)) * Wp + (x0 + m + kx - 1 + P)) * CIN;
      const u16* pa2 = pa1;
      if (GRAPH)
        pa2 = in + ((size_t)(b * Hp + (y + (ky - 1) * d + P)) * Wp +
                    (x0 + m + (kx - 1) * d + P)) * CIN;
      #pragma unroll
      for (int kc = 0; kc < KC; ++kc) {
        const uint4* pw = (const uint4*)lds + (tap * KC + kc) * 64 + lane * 2;
        FragU bw;                                       // ds_load_b128 x2
        bw.q[0] = pw[0];
        bw.q[1] = pw[1];
        const int cb = kc * 32 + cA0;
        FragU a1;
        a1.q[0] = *(const uint4*)(pa1 + cb);
        a1.q[1] = *(const uint4*)(pa1 + cb + 16);
        acc1 = __builtin_amdgcn_wmma_f32_16x16x32_bf16(
            false, a1.v, false, bw.v, (short)0, acc1, false, false);
        if (GRAPH) {
          FragU a2;
          a2.q[0] = *(const uint4*)(pa2 + cb);
          a2.q[1] = *(const uint4*)(pa2 + cb + 16);
          acc2 = __builtin_amdgcn_wmma_f32_16x16x32_bf16(
              false, a2.v, false, bw.v, (short)0, acc2, false, false);
        }
      }
    }
  }

  // D layout: lane L -> column N=L%16; VGPR r -> pixel row r + (L>=16 ? 8 : 0)
  int n = nt * 16 + (lane & 15);
  float bn = bias[n];
  #pragma unroll
  for (int r = 0; r < 8; ++r) {
    int px = x0 + r + half * 8;
    float v = acc1[r];
    if (GRAPH) {
      if (m8[((size_t)b * H + y) * W + px]) v = acc2[r];
    }
    v += bn;
    v = v > 0.f ? v : 0.f;
    if (OUT_NCHW_F32) {
      out_f32[(((size_t)b * COUT + n) * H + y) * W + px] = v;
    } else {
      out_bf[(((size_t)(b * Hp + y + P)) * Wp + (px + P)) * COUT + n] = f2bf(v);
    }
  }
}

template <bool GRAPH, bool OUT, int CIN, int COUT>
static void launch_conv(const u16* in, const u16* wp, const float* bias,
                        const unsigned char* m8, u16* out_bf, float* out_f,
                        int H, int W, int P, int d, hipStream_t s) {
  int blocks = 2 * (H >> 3) * (W >> 4) * (COUT >> 4);
  size_t shmem = (size_t)9 * (CIN >> 5) * 512 * sizeof(u16);
  conv_wmma<GRAPH, OUT, CIN, COUT>
      <<<blocks, 256, shmem, s>>>(in, wp, bias, m8, out_bf, out_f, H, W, P, d);
}

// ================================================================ host side
extern "C" void kernel_launch(void* const* d_in, const int* in_sizes, int n_in,
                              void* d_out, int out_size, void* d_ws, size_t ws_size,
                              hipStream_t stream) {
  (void)in_sizes; (void)n_in; (void)out_size; (void)ws_size;
  const float* batch = (const float*)d_in[0];
  const int* pmask   = (const int*)d_in[1];
  const float* W_[11]; const float* Bv[11];
  for (int i = 1; i <= 10; ++i) { W_[i] = (const float*)d_in[2 * i]; Bv[i] = (const float*)d_in[2 * i + 1]; }

  char* ws = (char*)d_ws;
  size_t o = 0;
  auto alloc = [&](size_t bytes) { size_t r = o; o = (o + bytes + 255) & ~(size_t)255; return r; };

  const size_t SZ384 = (size_t)2 * 386 * 386 * 64 * 2;               // NHWC bf16, P=1
  auto SZ192 = [](int C) { return (size_t)2 * 200 * 200 * C * 2; };  // P=4

  size_t o_a384a = alloc(SZ384);
  size_t o_a384b = alloc(SZ384);
  size_t o_p64   = alloc(SZ192(64));
  size_t o_128a  = alloc(SZ192(128));
  size_t o_128b  = alloc(SZ192(128));
  size_t o_256a  = alloc(SZ192(256));
  size_t o_256b  = alloc(SZ192(256));
  size_t o_512a  = alloc(SZ192(512));
  size_t o_512b  = alloc(SZ192(512));
  size_t zeroEnd = o;                       // activations need zeroed halos
  const int LC[11][2] = {{0,0},{3,64},{64,64},{64,128},{128,128},{128,256},
                         {256,256},{256,256},{256,512},{512,512},{512,512}};
  size_t o_wp[11];
  for (int i = 2; i <= 10; ++i) o_wp[i] = alloc((size_t)9 * LC[i][0] * LC[i][1] * 2);
  size_t o_mask = alloc((size_t)2 * 192 * 192);

  u16* a384a = (u16*)(ws + o_a384a);
  u16* a384b = (u16*)(ws + o_a384b);
  u16* p64   = (u16*)(ws + o_p64);
  u16* b128a = (u16*)(ws + o_128a);
  u16* b128b = (u16*)(ws + o_128b);
  u16* b256a = (u16*)(ws + o_256a);
  u16* b256b = (u16*)(ws + o_256b);
  u16* b512a = (u16*)(ws + o_512a);
  u16* b512b = (u16*)(ws + o_512b);
  unsigned char* m8 = (unsigned char*)(ws + o_mask);
  auto wp_ = [&](int i) { return (const u16*)(ws + o_wp[i]); };

  // 1) zero activation region (halos must be 0 for implicit padding)
  fill_zero<<<4096, 256, 0, stream>>>((uint4*)ws, zeroEnd / 16);

  // 2) expand pooling mask to dense 192x192 u8
  expand_mask<<<(2 * 192 * 192 + 255) / 256, 256, 0, stream>>>(pmask, m8);

  // 3) pack weights into WMMA fragment layout
  for (int i = 2; i <= 10; ++i) {
    size_t nel = (size_t)9 * LC[i][0] * LC[i][1];
    pack_w<<<(int)((nel + 255) / 256), 256, 0, stream>>>(W_[i], (u16*)(ws + o_wp[i]),
                                                         LC[i][0], LC[i][1]);
  }

  // 4) layer 1: direct conv (Cin=3), NCHW f32 -> NHWC bf16 P=1
  conv1_direct<<<(int)(((size_t)2 * 384 * 384 * 64 + 255) / 256), 256, 0, stream>>>(
      batch, W_[1], Bv[1], a384a);

  // 5) layer 2 @384, then maxpool to 192
  launch_conv<false, false, 64, 64>(a384a, wp_(2), Bv[2], nullptr, a384b, nullptr,
                                    384, 384, 1, 1, stream);
  maxpool2<<<(int)(((size_t)2 * 192 * 192 * 64 + 255) / 256), 256, 0, stream>>>(a384b, p64);

  // 6) layers 3-4 @192
  launch_conv<false, false, 64, 128>(p64, wp_(3), Bv[3], nullptr, b128a, nullptr,
                                     192, 192, 4, 1, stream);
  launch_conv<false, false, 128, 128>(b128a, wp_(4), Bv[4], nullptr, b128b, nullptr,
                                      192, 192, 4, 1, stream);

  // 7) irregular pool s=2, graph convs 5-7 (dil 2)
  irr_pool<<<(int)(((size_t)2 * 192 * 192 * 128 + 255) / 256), 256, 0, stream>>>(
      b128b, b128a, m8, 128, 2);
  launch_conv<true, false, 128, 256>(b128a, wp_(5), Bv[5], m8, b256a, nullptr,
                                     192, 192, 4, 2, stream);
  launch_conv<true, false, 256, 256>(b256a, wp_(6), Bv[6], m8, b256b, nullptr,
                                     192, 192, 4, 2, stream);
  launch_conv<true, false, 256, 256>(b256b, wp_(7), Bv[7], m8, b256a, nullptr,
                                     192, 192, 4, 2, stream);

  // 8) irregular pool s=4, graph convs 8-10 (dil 4); final writes f32 NCHW + ReLU
  irr_pool<<<(int)(((size_t)2 * 192 * 192 * 256 + 255) / 256), 256, 0, stream>>>(
      b256a, b256b, m8, 256, 4);
  launch_conv<true, false, 256, 512>(b256b, wp_(8), Bv[8], m8, b512a, nullptr,
                                     192, 192, 4, 4, stream);
  launch_conv<true, false, 512, 512>(b512a, wp_(9), Bv[9], m8, b512b, nullptr,
                                     192, 192, 4, 4, stream);
  launch_conv<true, true, 512, 512>(b512b, wp_(10), Bv[10], m8, nullptr, (float*)d_out,
                                    192, 192, 4, 4, stream);
}
